// MyLSTM_9672266351262
// MI455X (gfx1250) — matrix operand order repaired and checked by reading, not compiled, not run on hardware
//
#include <hip/hip_runtime.h>
#include <hip/hip_bf16.h>
#include <stdint.h>

// ---------------------------------------------------------------------------
// LSTM for MI455X (gfx1250): bf16 WMMA GEMMs + per-step recurrent kernel.
//   inputs: X[B,S,D] f32, W[D,4U] f32, U[D,4U] f32, b[4U] f32
//   output: concat( outputs[B,S,U], ht[B,U], ct[B,U] ) f32
// ---------------------------------------------------------------------------

typedef __attribute__((ext_vector_type(16))) __bf16   v16bf;
typedef __attribute__((ext_vector_type(8)))  float    v8f;
typedef __attribute__((ext_vector_type(4)))  unsigned uint4v;
typedef __attribute__((ext_vector_type(4)))  int      int4v;

#define BATCH 64
#define SEQ   512
#define DIM   1024
#define G4    4096                 // 4 * units
#define ROWS  (BATCH * SEQ)        // 32768 flattened (b,s) rows

struct Frag32B { uint4v lo, hi; }; // 32 bytes == one v16bf operand

__device__ __forceinline__ v16bf load_frag(const unsigned short* p0,
                                           const unsigned short* p1) {
    Frag32B f;
    f.lo = *(const uint4v*)p0;     // 16B load (global_load_b128 or ds_load_b128)
    f.hi = *(const uint4v*)p1;
    return __builtin_bit_cast(v16bf, f);
}

__device__ __forceinline__ unsigned short f2bf(float x) {
    unsigned u = __builtin_bit_cast(unsigned, x);
    u += 0x7FFFu + ((u >> 16) & 1u);           // round-to-nearest-even
    return (unsigned short)(u >> 16);
}

__device__ __forceinline__ float sigmoidf_fast(float x) {
    return 1.0f / (1.0f + __expf(-x));
}
__device__ __forceinline__ float tanhf_fast(float x) {
    return 2.0f / (1.0f + __expf(-2.0f * x)) - 1.0f;
}

// ---- async global->LDS staging (CDNA5 ASYNCcnt path), with safe fallback ---

__device__ __forceinline__ void async_copy_b128(const unsigned short* gsrc,
                                                unsigned short* ldst) {
#if __has_builtin(__builtin_amdgcn_global_load_async_to_lds_b128)
    __builtin_amdgcn_global_load_async_to_lds_b128(
        (__attribute__((address_space(1))) int4v*)(void*)gsrc,
        (__attribute__((address_space(3))) int4v*)(void*)ldst,
        0, 0);
#else
    *(uint4v*)ldst = *(const uint4v*)gsrc;     // sync fallback: b128 + ds_store
#endif
}

__device__ __forceinline__ void wait_async_copies() {
#if __has_builtin(__builtin_amdgcn_global_load_async_to_lds_b128)
#if __has_builtin(__builtin_amdgcn_s_wait_asynccnt)
    __builtin_amdgcn_s_wait_asynccnt(0);
#else
    asm volatile("s_wait_asynccnt 0x0" ::: "memory");
#endif
#endif
}

// --------------------------- prep kernels ----------------------------------

__global__ void cvt_f32_bf16(const float* __restrict__ in,
                             unsigned short* __restrict__ out, int n) {
    int i = blockIdx.x * blockDim.x + threadIdx.x;
    if (i < n) out[i] = f2bf(in[i]);
}

// in: [K,N] f32 row-major  ->  out: [N,K] bf16 (transposed, K contiguous)
__global__ void cvt_transpose(const float* __restrict__ in,
                              unsigned short* __restrict__ out,
                              int K, int N) {
    int i = blockIdx.x * blockDim.x + threadIdx.x;
    if (i >= K * N) return;
    int n = i / K;
    int k = i % K;
    out[i] = f2bf(in[(long long)k * N + n]);
}

__global__ void init_state(unsigned short* h0, unsigned short* h1, float* c) {
    int i = blockIdx.x * blockDim.x + threadIdx.x;
    if (i < BATCH * DIM) { h0[i] = 0; h1[i] = 0; c[i] = 0.0f; }
}

// --------------------------- phase 1: x_proj -------------------------------
// Block: 8 waves, 64 A-rows staged in LDS (128 KB). Wave w owns N-tile
// (blockIdx.y*8+w); each B-fragment load feeds 4 WMMAs (4 M-tiles) for
// 4x B reuse. A-panel staged once per block via async-to-LDS.

__global__ __launch_bounds__(256)
void gemm_xproj(const unsigned short* __restrict__ Xb,   // [ROWS, DIM] bf16
                const unsigned short* __restrict__ Wt,   // [G4, DIM]  bf16 (W^T)
                const float* __restrict__ bias,          // [G4]
                float* __restrict__ xproj)               // [ROWS, G4] f32
{
    __shared__ __align__(32) unsigned short As[64 * DIM];   // 128 KB

    const int tid   = threadIdx.x;
    const int lane  = tid & 31;
    const int wave  = tid >> 5;
    const int mbase = blockIdx.x * 64;
    const int n0    = (blockIdx.y * 8 + wave) * 16;

    // stage 64x1024 bf16 A panel (contiguous 128 KB): 256 thr * 16B * 32 iters
    {
        const unsigned short* gsrc = Xb + (long long)mbase * DIM;
        #pragma unroll
        for (int i = 0; i < 32; ++i) {
            const int idx8 = (i * 256 + tid) * 8;           // in halves
            async_copy_b128(gsrc + idx8, As + idx8);
        }
        wait_async_copies();
    }
    __syncthreads();

    const int col   = n0 + (lane & 15);
    const int akoff = (lane & 16) ? 8 : 0;    // A layout: lanes16-31 hold K+8.., K+24..
    const int bkoff = (lane & 16) ? 8 : 0;    // B layout: lanes16-31 hold K+8.., K+24..
    const unsigned short* Brow = Wt + (long long)col * DIM;
    const unsigned short* A0   = As + (lane & 15) * DIM;

    v8f acc0 = {}, acc1 = {}, acc2 = {}, acc3 = {};
    for (int kb = 0; kb < DIM; kb += 32) {
        __builtin_prefetch(Brow + kb + 64, 0, 1);           // global_prefetch_b8
        v16bf bm = load_frag(Brow + kb + bkoff, Brow + kb + bkoff + 16);
        v16bf a0 = load_frag(A0 +  0 * DIM + kb + akoff, A0 +  0 * DIM + kb + akoff + 16);
        v16bf a1 = load_frag(A0 + 16 * DIM + kb + akoff, A0 + 16 * DIM + kb + akoff + 16);
        v16bf a2 = load_frag(A0 + 32 * DIM + kb + akoff, A0 + 32 * DIM + kb + akoff + 16);
        v16bf a3 = load_frag(A0 + 48 * DIM + kb + akoff, A0 + 48 * DIM + kb + akoff + 16);
        acc0 = __builtin_amdgcn_wmma_f32_16x16x32_bf16(false, a0, false, bm, (short)0, acc0, false, false);
        acc1 = __builtin_amdgcn_wmma_f32_16x16x32_bf16(false, a1, false, bm, (short)0, acc1, false, false);
        acc2 = __builtin_amdgcn_wmma_f32_16x16x32_bf16(false, a2, false, bm, (short)0, acc2, false, false);
        acc3 = __builtin_amdgcn_wmma_f32_16x16x32_bf16(false, a3, false, bm, (short)0, acc3, false, false);
    }

    const float bv = bias[col];
    const int joff = (lane & 16) ? 8 : 0;     // f32 C/D layout
    #pragma unroll
    for (int j = 0; j < 8; ++j) {
        const int rrow = j + joff;
        xproj[(long long)(mbase +  0 + rrow) * G4 + col] = acc0[j] + bv;
        xproj[(long long)(mbase + 16 + rrow) * G4 + col] = acc1[j] + bv;
        xproj[(long long)(mbase + 32 + rrow) * G4 + col] = acc2[j] + bv;
        xproj[(long long)(mbase + 48 + rrow) * G4 + col] = acc3[j] + bv;
    }
}

// --------------------------- phase 2: recurrent step -----------------------
// Grid (4,64): block owns (row-tile r, unit-tile u0). h-tile (16x1024, 32 KB)
// async-staged into LDS once, then 4 waves (one per gate) run K=1024 WMMA
// chains reading A from LDS and U columns from the L2-resident global stream.

__global__ __launch_bounds__(128)
void lstm_step(const float* __restrict__ xproj,          // [ROWS, G4]
               const unsigned short* __restrict__ Ut,    // [G4, DIM] bf16 (U^T)
               const unsigned short* __restrict__ h_in,  // [BATCH, DIM] bf16
               unsigned short* __restrict__ h_out,       // [BATCH, DIM] bf16
               float* __restrict__ c,                    // [BATCH, DIM] f32
               float* __restrict__ out,                  // [BATCH, SEQ, DIM] f32
               int t)
{
    __shared__ __align__(32) unsigned short hA[16 * DIM];   // 32 KB
    __shared__ float zs[4][8][32];

    const int tid  = threadIdx.x;
    const int lane = tid & 31;
    const int gate = tid >> 5;                   // 0..3 = i,f,c,o
    const int r    = blockIdx.x;                 // batch row-tile 0..3
    const int u0   = blockIdx.y * 16;            // unit tile base

    // stage 16x1024 bf16 h panel (contiguous 32 KB): 128 thr * 16B * 16 iters
    {
        const unsigned short* gsrc = h_in + r * 16 * DIM;
        #pragma unroll
        for (int i = 0; i < 16; ++i) {
            const int idx8 = (i * 128 + tid) * 8;
            async_copy_b128(gsrc + idx8, hA + idx8);
        }
        wait_async_copies();
    }
    __syncthreads();

    const int col   = gate * DIM + u0 + (lane & 15);
    const int akoff = (lane & 16) ? 8 : 0;
    const int bkoff = (lane & 16) ? 8 : 0;
    const unsigned short* Arow = hA + (lane & 15) * DIM;
    const unsigned short* Brow = Ut + (long long)col * DIM;

    v8f acc = {};
    #pragma unroll 4
    for (int kb = 0; kb < DIM; kb += 32) {
        __builtin_prefetch(Brow + kb + 64, 0, 1);           // global_prefetch_b8
        v16bf a  = load_frag(Arow + kb + akoff, Arow + kb + akoff + 16);  // ds_load_b128
        v16bf bm = load_frag(Brow + kb + bkoff, Brow + kb + bkoff + 16);
        acc = __builtin_amdgcn_wmma_f32_16x16x32_bf16(
            false, a, false, bm, (short)0, acc, false, false);
    }

    #pragma unroll
    for (int j = 0; j < 8; ++j) zs[gate][j][lane] = acc[j];
    __syncthreads();

    // 256 tile elements, 128 threads -> 2 each
    #pragma unroll
    for (int e = tid; e < 256; e += 128) {
        const int m = e >> 4, n = e & 15;
        const int j = m & 7;
        const int l = n + ((m & 8) << 1);        // invert f32 C/D VGPR layout
        float zi = zs[0][j][l];
        float zf = zs[1][j][l];
        float zc = zs[2][j][l];
        float zo = zs[3][j][l];

        const int bb   = r * 16 + m;
        const int unit = u0 + n;
        const long long xrow = (long long)(bb * SEQ + t) * G4;
        zi += xproj[xrow + 0 * DIM + unit];
        zf += xproj[xrow + 1 * DIM + unit];
        zc += xproj[xrow + 2 * DIM + unit];
        zo += xproj[xrow + 3 * DIM + unit];

        const float ig = sigmoidf_fast(zi);
        const float fg = sigmoidf_fast(zf);
        const float cg = tanhf_fast(zc);
        const float og = sigmoidf_fast(zo);

        const int cidx = bb * DIM + unit;
        const float cn = fg * c[cidx] + ig * cg;
        c[cidx] = cn;
        const float h = og * tanhf_fast(cn);
        out[(long long)(bb * SEQ + t) * DIM + unit] = h;
        h_out[cidx] = f2bf(h);
    }
}

// --------------------------- finalize: ht, ct ------------------------------

__global__ void finalize(const float* __restrict__ c, float* __restrict__ dout) {
    int i = blockIdx.x * blockDim.x + threadIdx.x;
    if (i >= BATCH * DIM) return;
    const int bb = i / DIM, u = i % DIM;
    const long long OUTN = (long long)BATCH * SEQ * DIM;
    dout[OUTN + i] = dout[(long long)(bb * SEQ + (SEQ - 1)) * DIM + u]; // ht
    dout[OUTN + BATCH * DIM + i] = c[i];                               // ct
}

// ---------------------------------------------------------------------------

extern "C" void kernel_launch(void* const* d_in, const int* in_sizes, int n_in,
                              void* d_out, int out_size, void* d_ws, size_t ws_size,
                              hipStream_t stream)
{
    const float* X    = (const float*)d_in[0];
    const float* W    = (const float*)d_in[1];
    const float* U    = (const float*)d_in[2];
    const float* bias = (const float*)d_in[3];
    float* out = (float*)d_out;

    // workspace layout
    char* ws = (char*)d_ws;
    unsigned short* Xb = (unsigned short*)ws;                         // 64 MB
    size_t off = (size_t)ROWS * DIM * 2;
    unsigned short* Wt = (unsigned short*)(ws + off); off += (size_t)G4 * DIM * 2; // 8 MB
    unsigned short* Ut = (unsigned short*)(ws + off); off += (size_t)G4 * DIM * 2; // 8 MB
    float* xproj = (float*)(ws + off);                off += (size_t)ROWS * G4 * 4; // 512 MB
    unsigned short* h0 = (unsigned short*)(ws + off); off += (size_t)BATCH * DIM * 2;
    unsigned short* h1 = (unsigned short*)(ws + off); off += (size_t)BATCH * DIM * 2;
    float* c = (float*)(ws + off);

    // precision conversion + weight transposes (K-contiguous for WMMA B loads)
    cvt_f32_bf16<<<(ROWS * DIM + 255) / 256, 256, 0, stream>>>(X, Xb, ROWS * DIM);
    cvt_transpose<<<(DIM * G4 + 255) / 256, 256, 0, stream>>>(W, Wt, DIM, G4);
    cvt_transpose<<<(DIM * G4 + 255) / 256, 256, 0, stream>>>(U, Ut, DIM, G4);
    init_state<<<(BATCH * DIM + 255) / 256, 256, 0, stream>>>(h0, h1, c);

    // phase 1: one big WMMA GEMM for all timesteps' input projections
    dim3 ggrid(ROWS / 64, G4 / (16 * 8));
    gemm_xproj<<<ggrid, 256, 0, stream>>>(Xb, Wt, bias, xproj);

    // phase 2: 512 serial recurrent steps (graph replay amortizes launches)
    dim3 sgrid(BATCH / 16, DIM / 16);
    for (int t = 0; t < SEQ; ++t) {
        const unsigned short* hi = (t & 1) ? h1 : h0;
        unsigned short*       ho = (t & 1) ? h0 : h1;
        lstm_step<<<sgrid, 128, 0, stream>>>(xproj, Ut, hi, ho, c, out, t);
    }

    finalize<<<(BATCH * DIM + 255) / 256, 256, 0, stream>>>(c, out);
}
